// AttentionLayer_86964497810051
// MI455X (gfx1250) — matrix operand order repaired and checked by reading, not compiled, not run on hardware
//
#include <hip/hip_runtime.h>
#include <hip/hip_bf16.h>

// ---------------------------------------------------------------------------
// CDNA5 (gfx1250) transformer encoder layer, bf16 WMMA + f32 accumulate.
// B=4, L=2048, D=512, H=8, Dh=64.
// ---------------------------------------------------------------------------

typedef __attribute__((ext_vector_type(16))) __bf16 v16bf;
typedef __attribute__((ext_vector_type(8)))  __bf16 v8bf;
typedef __attribute__((ext_vector_type(8)))  float  v8f;

#define TB 4
#define TL 2048
#define TD 512
#define TH 8
#define TDH 64
#define TM (TB * TL)          // 8192 rows of the token matrix

__device__ __forceinline__ v8f wmma_bf16(v16bf a, v16bf b, v8f c) {
    // v_wmma_f32_16x16x32_bf16
    return __builtin_amdgcn_wmma_f32_16x16x32_bf16(false, a, false, b,
                                                   (short)0, c, false, false);
}

// A-matrix 16x32 bf16 fragment (M x K). Per ISA: lanes 0-15 hold K 0..7 &
// 16..23, lanes 16-31 hold K 8..15 & 24..31, row M = lane & 15.
__device__ __forceinline__ v16bf load_fragA(const __bf16* base, int ld, int k0) {
    int lane = threadIdx.x & 31;
    int row  = lane & 15;
    int kb   = (lane >> 4) << 3;      // 0 or 8
    const __bf16* p0 = base + row * ld + k0 + kb;
    v8bf lo = *(const v8bf*)p0;       // K = k0+kb .. +7
    v8bf hi = *(const v8bf*)(p0 + 16);// K = k0+16+kb .. +7
    return __builtin_shufflevector(lo, hi, 0,1,2,3,4,5,6,7,8,9,10,11,12,13,14,15);
}

// B-matrix 32x16 bf16 fragment (K x N), column n = lane & 15, lanes 0-15 hold
// K 0..7 & 16..23, lanes 16-31 hold K 8..15 & 24..31. `base` points at
// row-major [N, K] storage (i.e. W stored as W[n][k]), so a B-column is a
// contiguous row of W.
__device__ __forceinline__ v16bf load_fragB(const __bf16* base, int ld) {
    int lane = threadIdx.x & 31;
    const __bf16* p = base + (lane & 15) * ld + ((lane >> 4) << 3);
    v8bf lo = *(const v8bf*)p;
    v8bf hi = *(const v8bf*)(p + 16);
    return __builtin_shufflevector(lo, hi, 0,1,2,3,4,5,6,7,8,9,10,11,12,13,14,15);
}

// ---------------------------------------------------------------------------
// fp32 -> bf16 conversion
// ---------------------------------------------------------------------------
__global__ __launch_bounds__(256) void cvt_bf16_kernel(const float* __restrict__ src,
                                                       __bf16* __restrict__ dst, int n) {
    for (int i = blockIdx.x * blockDim.x + threadIdx.x; i < n;
         i += gridDim.x * blockDim.x)
        dst[i] = (__bf16)src[i];
}

// ---------------------------------------------------------------------------
// Generic GEMM: C[m,n] = sum_k A[m,k] * W[n,k]  (A: [M,512] bf16, W: [512,512] bf16)
// Block = 256 threads (8 waves), computes 16 rows x all 512 columns.
// Wave w owns N columns [w*64, w*64+64) = 4 WMMA N-tiles; 16 K-steps of 32.
// Store modes:
//   0: Q/K head layout  out_bf[((b*8+h)*2048+l)*64+d] = v*scale
//   1: V transposed     out_bf[((b*8+h)*64+d)*2048+l] = v
//   2: bias+ReLU        out_bf[m*512+n] = max(v+bias[n],0)
//   3: bias, fp32       out_f32[m*512+n] = v+bias[n]
// ---------------------------------------------------------------------------
__global__ __launch_bounds__(256) void gemm_wmma_kernel(const __bf16* __restrict__ A,
                                                        const __bf16* __restrict__ W,
                                                        const float* __restrict__ bias,
                                                        void* __restrict__ out,
                                                        int mode, float scale) {
    __shared__ __bf16 sA[16 * TD];            // 16 KB A tile
    const int m0 = blockIdx.x * 16;

    // Stage the 16x512 A tile into LDS (contiguous b128 copies).
    {
        const v8bf* src = (const v8bf*)(A + (size_t)m0 * TD);
        v8bf* dst = (v8bf*)sA;
        for (int i = threadIdx.x; i < 16 * TD / 8; i += 256) dst[i] = src[i];
    }
    __syncthreads();

    const int wave = threadIdx.x >> 5;
    v8f acc[4] = {};

    for (int ks = 0; ks < 16; ++ks) {
        v16bf a = load_fragA(sA, TD, ks * 32);
#pragma unroll
        for (int j = 0; j < 4; ++j) {
            const __bf16* wb = W + (size_t)(wave * 64 + j * 16) * TD + ks * 32;
            if (ks + 1 < 16) __builtin_prefetch(wb + 32, 0, 0);   // global_prefetch_b8
            acc[j] = wmma_bf16(a, load_fragB(wb, TD), acc[j]);
        }
    }

    // Epilogue: C/D layout M = ((lane>>4)<<3)+i, N = lane&15.
    const int lane = threadIdx.x & 31;
    const int nl = lane & 15;
    const int mb = (lane >> 4) << 3;
#pragma unroll
    for (int j = 0; j < 4; ++j) {
        const int n = wave * 64 + j * 16 + nl;
#pragma unroll
        for (int i = 0; i < 8; ++i) {
            const int m = m0 + mb + i;
            float v = acc[j][i];
            if (mode == 0) {
                int b = m >> 11, l = m & 2047, h = n >> 6, d = n & 63;
                ((__bf16*)out)[(((size_t)(b * 8 + h)) * TL + l) * TDH + d] =
                    (__bf16)(v * scale);
            } else if (mode == 1) {
                int b = m >> 11, l = m & 2047, h = n >> 6, d = n & 63;
                ((__bf16*)out)[(((size_t)(b * 8 + h)) * TDH + d) * TL + l] = (__bf16)v;
            } else if (mode == 2) {
                v = fmaxf(v + bias[n], 0.0f);
                ((__bf16*)out)[(size_t)m * TD + n] = (__bf16)v;
            } else {
                ((float*)out)[(size_t)m * TD + n] = v + bias[n];
            }
        }
    }
}

// ---------------------------------------------------------------------------
// Attention: per (b,h) and 16-row Q tile.
//   S = Q_tile[16x64] @ K^T  -> LDS (16 x 2048 f32), softmax in LDS,
//   attn written fp32 to d_out, then ctx = P[16x2048] @ V[2048x64] with the
//   K dimension split across the 8 waves, reduced via ds_add_f32.
// Dynamic LDS: 16*2048 + 1024 + 256 floats = 136192 bytes (WGP has 320 KB).
// ---------------------------------------------------------------------------
__global__ __launch_bounds__(256) void attn_kernel(const __bf16* __restrict__ Q,
                                                   const __bf16* __restrict__ K,
                                                   const __bf16* __restrict__ VT,
                                                   float* __restrict__ attn,
                                                   float* __restrict__ ctx) {
    extern __shared__ float smem[];
    float* sS   = smem;                 // [16][2048] scores -> probs
    float* sCtx = smem + 16 * TL;       // [16][64] ctx accumulator
    float* sRed = sCtx + 16 * TDH;      // [16][16] reduction scratch

    const int bh = blockIdx.y;          // b*8 + h
    const int m0 = blockIdx.x * 16;
    const int wave = threadIdx.x >> 5;
    const int lane = threadIdx.x & 31;
    const int nl = lane & 15;
    const int mb = (lane >> 4) << 3;

    const __bf16* qbase = Q + ((size_t)bh * TL + m0) * TDH;
    const __bf16* kbase = K + (size_t)bh * TL * TDH;

    // ---- Phase 1: scores. Q is pre-scaled by 1/sqrt(Dh). -------------------
    v16bf a0 = load_fragA(qbase, TDH, 0);
    v16bf a1 = load_fragA(qbase, TDH, 32);
    for (int ct = wave; ct < TL / 16; ct += 8) {
        v8f acc = {};
        const __bf16* kb = kbase + (size_t)ct * 16 * TDH;
        acc = wmma_bf16(a0, load_fragB(kb, TDH), acc);
        acc = wmma_bf16(a1, load_fragB(kb + 32, TDH), acc);
        const int n = ct * 16 + nl;
#pragma unroll
        for (int i = 0; i < 8; ++i) sS[(mb + i) * TL + n] = acc[i];
    }
    __syncthreads();

    // ---- Phase 2: softmax over each of the 16 rows (16 threads / row). ----
    const int row = threadIdx.x >> 4;
    const int c = threadIdx.x & 15;
    float* srow = sS + row * TL;

    float mx = -1e30f;
    for (int j = c; j < TL; j += 16) mx = fmaxf(mx, srow[j]);
    sRed[row * 16 + c] = mx;
    __syncthreads();
    if (c == 0) {
        float m2 = sRed[row * 16];
        for (int t = 1; t < 16; ++t) m2 = fmaxf(m2, sRed[row * 16 + t]);
        sRed[row * 16] = m2;
    }
    __syncthreads();
    mx = sRed[row * 16];

    float sum = 0.0f;
    for (int j = c; j < TL; j += 16) {
        float e = __expf(srow[j] - mx);
        srow[j] = e;
        sum += e;
    }
    __syncthreads();
    sRed[row * 16 + c] = sum;
    __syncthreads();
    if (c == 0) {
        float s2 = 0.0f;
        for (int t = 0; t < 16; ++t) s2 += sRed[row * 16 + t];
        sRed[row * 16] = s2;
    }
    __syncthreads();
    const float inv = 1.0f / sRed[row * 16];

    float* arow = attn + ((size_t)bh * TL + m0 + row) * TL;
    for (int j = c; j < TL; j += 16) {
        float p = srow[j] * inv;
        srow[j] = p;
        arow[j] = p;          // fp32 attention output (required by reference)
    }
    for (int i = threadIdx.x; i < 16 * TDH; i += 256) sCtx[i] = 0.0f;
    __syncthreads();

    // ---- Phase 3: ctx = P @ V. Wave w handles K in [w*256, w*256+256). ----
    v8f acc[4] = {};
    const int k0w = wave * 256;
    for (int ks = 0; ks < 8; ++ks) {
        const int k0 = k0w + ks * 32;
        const int arow_m = lane & 15;
        const int kb = (lane >> 4) << 3;
        const float* sp = sS + arow_m * TL + k0;
        v16bf a;
#pragma unroll
        for (int i = 0; i < 8; ++i) {
            a[i]     = (__bf16)sp[kb + i];
            a[8 + i] = (__bf16)sp[16 + kb + i];
        }
#pragma unroll
        for (int nt = 0; nt < 4; ++nt) {
            const __bf16* vb = VT + ((size_t)bh * TDH + nt * 16) * TL + k0;
            acc[nt] = wmma_bf16(a, load_fragB(vb, TL), acc[nt]);
        }
    }
#pragma unroll
    for (int nt = 0; nt < 4; ++nt)
#pragma unroll
        for (int i = 0; i < 8; ++i)
            atomicAdd(&sCtx[(mb + i) * TDH + nt * 16 + nl], acc[nt][i]);  // ds_add_f32
    __syncthreads();

    // Scatter ctx into [B, L, D] layout (head-interleaved columns).
    const int b = bh >> 3, h = bh & 7;
    for (int i = threadIdx.x; i < 16 * TDH; i += 256) {
        int m = i >> 6, d = i & 63;
        ctx[((size_t)(b * TL + m0 + m)) * TD + h * TDH + d] = sCtx[i];
    }
}

// ---------------------------------------------------------------------------
// Residual + LayerNorm over rows of 512. One block per row, 2 elems/thread.
// Optionally emits a bf16 copy for the following GEMM.
// ---------------------------------------------------------------------------
__global__ __launch_bounds__(256) void ln_res_kernel(const float* __restrict__ x,
                                                     const float* __restrict__ res,
                                                     const float* __restrict__ g,
                                                     const float* __restrict__ bta,
                                                     float* __restrict__ outF,
                                                     __bf16* __restrict__ outBf) {
    __shared__ float rsum[256];
    __shared__ float rsq[256];
    const size_t base = (size_t)blockIdx.x * TD;
    const int t = threadIdx.x;

    float v0 = x[base + t]       + res[base + t];
    float v1 = x[base + 256 + t] + res[base + 256 + t];
    rsum[t] = v0 + v1;
    rsq[t]  = v0 * v0 + v1 * v1;
    __syncthreads();
    for (int s = 128; s > 0; s >>= 1) {
        if (t < s) { rsum[t] += rsum[t + s]; rsq[t] += rsq[t + s]; }
        __syncthreads();
    }
    const float mean = rsum[0] * (1.0f / TD);
    const float var  = rsq[0] * (1.0f / TD) - mean * mean;
    const float rstd = rsqrtf(var + 1e-6f);

    float o0 = (v0 - mean) * rstd * g[t] + bta[t];
    float o1 = (v1 - mean) * rstd * g[t + 256] + bta[t + 256];
    outF[base + t] = o0;
    outF[base + 256 + t] = o1;
    if (outBf) {
        outBf[base + t] = (__bf16)o0;
        outBf[base + 256 + t] = (__bf16)o1;
    }
}

// ---------------------------------------------------------------------------
// Host launch
// ---------------------------------------------------------------------------
extern "C" void kernel_launch(void* const* d_in, const int* in_sizes, int n_in,
                              void* d_out, int out_size, void* d_ws, size_t ws_size,
                              hipStream_t stream) {
    (void)in_sizes; (void)n_in; (void)out_size;

    const float* inp  = (const float*)d_in[0];
    const float* wq   = (const float*)d_in[1];
    const float* wk   = (const float*)d_in[2];
    const float* wv   = (const float*)d_in[3];
    const float* ln1g = (const float*)d_in[4];
    const float* ln1b = (const float*)d_in[5];
    const float* w1   = (const float*)d_in[6];
    const float* b1   = (const float*)d_in[7];
    const float* w2   = (const float*)d_in[8];
    const float* b2   = (const float*)d_in[9];
    const float* ln2g = (const float*)d_in[10];
    const float* ln2b = (const float*)d_in[11];

    float* out_p  = (float*)d_out;
    float* attn_p = out_p + (size_t)TB * TL * TD;   // tuple order: (out, attn)

    // Bump allocator on d_ws (256 B aligned).
    size_t off = 0;
    auto alloc = [&](size_t bytes) {
        void* p = (char*)d_ws + off;
        off += (bytes + 255) & ~(size_t)255;
        return p;
    };
    const size_t actN = (size_t)TM * TD;   // 8192*512
    const size_t wN   = (size_t)TD * TD;   // 512*512

    __bf16* x_bf   = (__bf16*)alloc(actN * 2);
    __bf16* wq_bf  = (__bf16*)alloc(wN * 2);
    __bf16* wk_bf  = (__bf16*)alloc(wN * 2);
    __bf16* wv_bf  = (__bf16*)alloc(wN * 2);
    __bf16* w1_bf  = (__bf16*)alloc(wN * 2);
    __bf16* w2_bf  = (__bf16*)alloc(wN * 2);
    __bf16* q_bf   = (__bf16*)alloc(actN * 2);   // [B,H,L,Dh]
    __bf16* k_bf   = (__bf16*)alloc(actN * 2);   // [B,H,L,Dh]
    __bf16* vT_bf  = (__bf16*)alloc(actN * 2);   // [B,H,Dh,L]
    float*  ctx    = (float*)alloc(actN * 4);    // [B,L,D]
    float*  hid    = (float*)alloc(actN * 4);
    __bf16* hid_bf = (__bf16*)alloc(actN * 2);
    __bf16* act_bf = (__bf16*)alloc(actN * 2);
    float*  ff     = (float*)alloc(actN * 4);
    (void)ws_size;

    // 1) fp32 -> bf16 conversions.
    cvt_bf16_kernel<<<1024, 256, 0, stream>>>(inp, x_bf, (int)actN);
    cvt_bf16_kernel<<<256, 256, 0, stream>>>(wq, wq_bf, (int)wN);
    cvt_bf16_kernel<<<256, 256, 0, stream>>>(wk, wk_bf, (int)wN);
    cvt_bf16_kernel<<<256, 256, 0, stream>>>(wv, wv_bf, (int)wN);
    cvt_bf16_kernel<<<256, 256, 0, stream>>>(w1, w1_bf, (int)wN);
    cvt_bf16_kernel<<<256, 256, 0, stream>>>(w2, w2_bf, (int)wN);

    const int gemmBlocks = TM / 16;   // 512
    // 2) QKV projections (softmax scale folded into Q).
    gemm_wmma_kernel<<<gemmBlocks, 256, 0, stream>>>(x_bf, wq_bf, nullptr, q_bf, 0, 0.125f);
    gemm_wmma_kernel<<<gemmBlocks, 256, 0, stream>>>(x_bf, wk_bf, nullptr, k_bf, 0, 1.0f);
    gemm_wmma_kernel<<<gemmBlocks, 256, 0, stream>>>(x_bf, wv_bf, nullptr, vT_bf, 1, 1.0f);

    // 3) Attention (scores + softmax + attn output + ctx).
    const size_t attnSmem = (size_t)(16 * TL + 16 * TDH + 256) * sizeof(float); // 136192 B
    attn_kernel<<<dim3(TL / 16, TB * TH), 256, attnSmem, stream>>>(q_bf, k_bf, vT_bf,
                                                                   attn_p, ctx);

    // 4) hid = LN1(ctx + inp)   (+ bf16 copy for FFN1).
    ln_res_kernel<<<TM, 256, 0, stream>>>(ctx, inp, ln1g, ln1b, hid, hid_bf);

    // 5) FFN: relu(hid@w1^T + b1) @ w2^T + b2.
    gemm_wmma_kernel<<<gemmBlocks, 256, 0, stream>>>(hid_bf, w1_bf, b1, act_bf, 2, 1.0f);
    gemm_wmma_kernel<<<gemmBlocks, 256, 0, stream>>>(act_bf, w2_bf, b2, ff, 3, 1.0f);

    // 6) out = LN2(ff + hid).
    ln_res_kernel<<<TM, 256, 0, stream>>>(ff, hid, ln2g, ln2b, out_p, nullptr);
}
